// GCN_11905649344775
// MI455X (gfx1250) — compile-verified, matrix operand order checked by reading, not run on hardware
//
#include <hip/hip_runtime.h>
#include <hip/hip_bf16.h>

// ---------------------------------------------------------------------------
// Types for CDNA5 WMMA (wave32): V_WMMA_F32_16X16X32_BF16
// ---------------------------------------------------------------------------
typedef __attribute__((ext_vector_type(16))) __bf16 v16bf;
typedef __attribute__((ext_vector_type(8)))  float  v8f;

__device__ __forceinline__ unsigned short f2bf(float x) {
    union { float f; unsigned int u; } v; v.f = x;
    unsigned int b = v.u + 0x7FFFu + ((v.u >> 16) & 1u);   // round-to-nearest-even
    return (unsigned short)(b >> 16);
}

union FragBF { uint4 q[2]; v16bf v; };

// Load one 16x32-bf16 operand fragment from an LDS row (row stride = 48
// ushorts = 96B, 16B aligned). Lane-group g selects K slices:
//   bytes [16g, 16g+16) -> K = 8g..8g+7   and  [32+16g, ..) -> K = 16+8g..
__device__ __forceinline__ v16bf load_frag(const unsigned short* row, int g) {
    FragBF f;
    const uint4* p = reinterpret_cast<const uint4*>(row);
    f.q[0] = p[g];
    f.q[1] = p[2 + g];
    return f.v;
}

// ---------------------------------------------------------------------------
// Tiled bf16 WMMA GEMM:  C[M,Nc] = A[M,K] @ B[K,Nc] + bias (+optional relu)
// Block = 256 threads (8 waves). Tile 128x128, BK=32.
// Requires: K % 32 == 0, Nc % 128 == 0 (true for all GEMMs here). M guarded.
// ---------------------------------------------------------------------------
#define LDS_STRIDE 48

__global__ __launch_bounds__(256)
void wmma_gemm_kernel(const float* __restrict__ A, const float* __restrict__ B,
                      const float* __restrict__ bias, float* __restrict__ C,
                      int M, int K, int Nc, int relu)
{
    __shared__ __align__(16) unsigned short As[128 * LDS_STRIDE];
    __shared__ __align__(16) unsigned short Bs[128 * LDS_STRIDE];

    const int tid  = threadIdx.x;
    const int lane = tid & 31;
    const int wave = tid >> 5;          // 0..7
    const int wm   = wave >> 1;         // 0..3 : 32-row slab
    const int wn   = wave & 1;          // 0..1 : 64-col slab
    const int tileM = blockIdx.y * 128;
    const int tileN = blockIdx.x * 128;

    const int lm  = lane & 15;
    const int grp = lane >> 4;

    v8f zero8 = {0.f,0.f,0.f,0.f,0.f,0.f,0.f,0.f};
    v8f acc[2][4];
    #pragma unroll
    for (int i = 0; i < 2; ++i)
        #pragma unroll
        for (int j = 0; j < 4; ++j)
            acc[i][j] = zero8;

    // cooperative-load mappings
    const int ar = tid >> 1;            // 0..127 : A row in tile
    const int ac = (tid & 1) * 16;      // 0 / 16 : A col base
    const int am = tileM + ar;
    const bool avalid = (am < M);
    const int bk = tid >> 3;            // 0..31  : B k in tile
    const int bn = (tid & 7) * 16;      // 0..112 : B col base

    for (int k0 = 0; k0 < K; k0 += 32) {
        // ---- stage A (row-major, fp32 -> bf16) ----
        {
            const float* ap = A + (long long)am * K + k0 + ac;
            #pragma unroll
            for (int v = 0; v < 4; ++v) {
                float4 d = avalid ? *(const float4*)(ap + v * 4)
                                  : make_float4(0.f, 0.f, 0.f, 0.f);
                unsigned short* s = &As[ar * LDS_STRIDE + ac + v * 4];
                s[0] = f2bf(d.x); s[1] = f2bf(d.y);
                s[2] = f2bf(d.z); s[3] = f2bf(d.w);
            }
        }
        // ---- stage B transposed: Bs[n][k] ----
        {
            const float* bp = B + (long long)(k0 + bk) * Nc + tileN + bn;
            #pragma unroll
            for (int v = 0; v < 4; ++v) {
                float4 d = *(const float4*)(bp + v * 4);
                Bs[(bn + v*4 + 0) * LDS_STRIDE + bk] = f2bf(d.x);
                Bs[(bn + v*4 + 1) * LDS_STRIDE + bk] = f2bf(d.y);
                Bs[(bn + v*4 + 2) * LDS_STRIDE + bk] = f2bf(d.z);
                Bs[(bn + v*4 + 3) * LDS_STRIDE + bk] = f2bf(d.w);
            }
        }
        __syncthreads();

        v16bf afrag[2];
        #pragma unroll
        for (int i = 0; i < 2; ++i)
            afrag[i] = load_frag(&As[(wm * 32 + i * 16 + lm) * LDS_STRIDE], grp);

        #pragma unroll
        for (int j = 0; j < 4; ++j) {
            v16bf bfrag = load_frag(&Bs[(wn * 64 + j * 16 + lm) * LDS_STRIDE], grp);
            #pragma unroll
            for (int i = 0; i < 2; ++i) {
                acc[i][j] = __builtin_amdgcn_wmma_f32_16x16x32_bf16(
                    false, afrag[i], false, bfrag, (short)0, acc[i][j],
                    false, false);
            }
        }
        __syncthreads();
    }

    // ---- epilogue: C/D layout -> lanes 0-15: M=r, lanes 16-31: M=r+8 ----
    #pragma unroll
    for (int i = 0; i < 2; ++i) {
        const int mrow = tileM + wm * 32 + i * 16 + grp * 8;
        #pragma unroll
        for (int j = 0; j < 4; ++j) {
            const int ncol = tileN + wn * 64 + j * 16 + lm;
            const float bv = bias ? bias[ncol] : 0.0f;
            #pragma unroll
            for (int r = 0; r < 8; ++r) {
                int m = mrow + r;
                if (m < M) {
                    float x = acc[i][j][r] + bv;
                    if (relu) x = fmaxf(x, 0.0f);
                    C[(long long)m * Nc + ncol] = x;
                }
            }
        }
    }
}

// ---------------------------------------------------------------------------
// Elementwise / graph kernels (fp32 exact)
// ---------------------------------------------------------------------------
__global__ void zero_kernel(float* __restrict__ p, long long n4) {
    long long i = (long long)blockIdx.x * blockDim.x + threadIdx.x;
    if (i < n4) ((float4*)p)[i] = make_float4(0.f, 0.f, 0.f, 0.f);
}

// msg (holds e on entry) := relu(h[src] + e) + 1e-7   (in place)
__global__ void edge_message_kernel(float* __restrict__ msg,
                                    const float* __restrict__ h,
                                    const int* __restrict__ src,
                                    int E, int F) {
    int q = F >> 2;
    long long idx = (long long)blockIdx.x * blockDim.x + threadIdx.x;
    if (idx >= (long long)E * q) return;
    int ei = (int)(idx / q);
    int f  = (int)(idx - (long long)ei * q) << 2;
    int s  = src[ei];
    float4 hv = *(const float4*)(h + (long long)s * F + f);
    float4 ev = *(const float4*)(msg + (long long)ei * F + f);
    float4 r;
    r.x = fmaxf(hv.x + ev.x, 0.f) + 1e-7f;
    r.y = fmaxf(hv.y + ev.y, 0.f) + 1e-7f;
    r.z = fmaxf(hv.z + ev.z, 0.f) + 1e-7f;
    r.w = fmaxf(hv.w + ev.w, 0.f) + 1e-7f;
    *(float4*)(msg + (long long)ei * F + f) = r;
}

// segment max: all msg > 0, so uint bit-pattern order == float order; mx init 0
__global__ void segmax_kernel(unsigned int* __restrict__ mx,
                              const float* __restrict__ msg,
                              const int* __restrict__ dst, int E, int F) {
    long long idx = (long long)blockIdx.x * blockDim.x + threadIdx.x;
    if (idx >= (long long)E * F) return;
    int ei = (int)(idx / F);
    int f  = (int)(idx - (long long)ei * F);
    atomicMax(&mx[(long long)dst[ei] * F + f], __float_as_uint(msg[idx]));
}

__global__ void expden_kernel(float* __restrict__ den,
                              const float* __restrict__ msg,
                              const unsigned int* __restrict__ mx,
                              const int* __restrict__ dst, int E, int F) {
    long long idx = (long long)blockIdx.x * blockDim.x + threadIdx.x;
    if (idx >= (long long)E * F) return;
    int ei = (int)(idx / F);
    int f  = (int)(idx - (long long)ei * F);
    long long o = (long long)dst[ei] * F + f;
    float ex = __expf(msg[idx] - __uint_as_float(mx[o]));
    atomicAdd(&den[o], ex);
}

__global__ void agg_kernel(float* __restrict__ agg,
                           const float* __restrict__ msg,
                           const unsigned int* __restrict__ mx,
                           const float* __restrict__ den,
                           const int* __restrict__ dst, int E, int F) {
    long long idx = (long long)blockIdx.x * blockDim.x + threadIdx.x;
    if (idx >= (long long)E * F) return;
    int ei = (int)(idx / F);
    int f  = (int)(idx - (long long)ei * F);
    long long o = (long long)dst[ei] * F + f;
    float v  = msg[idx];
    float ex = __expf(v - __uint_as_float(mx[o]));
    atomicAdd(&agg[o], ex * v / (den[o] + 1e-16f));
}

__global__ void add_kernel(float* __restrict__ o, const float* __restrict__ a,
                           const float* __restrict__ b, long long n4) {
    long long i = (long long)blockIdx.x * blockDim.x + threadIdx.x;
    if (i >= n4) return;
    float4 av = ((const float4*)a)[i];
    float4 bv = ((const float4*)b)[i];
    ((float4*)o)[i] = make_float4(av.x + bv.x, av.y + bv.y, av.z + bv.z, av.w + bv.w);
}

// BatchNorm (train stats) over rows: coalesced per-block partial sums
__global__ void bn_partial_kernel(const float* __restrict__ h1,
                                  float* __restrict__ sums, int Nrows, int C) {
    int r0 = blockIdx.x * 256;
    int rend = r0 + 256; if (rend > Nrows) rend = Nrows;
    for (int c = threadIdx.x; c < C; c += 256) {
        float s = 0.f, q = 0.f;
        for (int r = r0; r < rend; ++r) {
            float v = h1[(long long)r * C + c];
            s += v; q += v * v;
        }
        atomicAdd(&sums[c], s);
        atomicAdd(&sums[C + c], q);
    }
}

__global__ void bn_finalize_kernel(const float* __restrict__ sums,
                                   float* __restrict__ stats, int Nrows, int C) {
    int c = blockIdx.x * blockDim.x + threadIdx.x;
    if (c >= C) return;
    float inv = 1.0f / (float)Nrows;
    float mu  = sums[c] * inv;
    float var = sums[C + c] * inv - mu * mu;
    stats[c]     = mu;
    stats[C + c] = rsqrtf(fmaxf(var, 0.f) + 1e-5f);
}

__global__ void bn_apply_relu_kernel(float* __restrict__ h1,
                                     const float* __restrict__ stats,
                                     const float* __restrict__ gam,
                                     const float* __restrict__ bet,
                                     long long total, int C) {
    long long idx = (long long)blockIdx.x * blockDim.x + threadIdx.x;
    if (idx >= total) return;
    int c = (int)(idx % C);
    float v = h1[idx];
    v = (v - stats[c]) * stats[C + c] * gam[c] + bet[c];
    h1[idx] = fmaxf(v, 0.f);
}

// ---------------------------------------------------------------------------
// Host-side orchestration
// ---------------------------------------------------------------------------
static inline void launch_gemm(const float* A, const float* B, const float* bias,
                               float* C, int M, int K, int Nc, int relu,
                               hipStream_t s) {
    dim3 grid(Nc / 128, (M + 127) / 128);
    wmma_gemm_kernel<<<grid, 256, 0, s>>>(A, B, bias, C, M, K, Nc, relu);
}

static void run_layer(const float* x, int Fi, int Fo,
                      const int* src, const int* dst, const float* ea,
                      const float* const* P, float* yout, int yrelu,
                      float* MSG, float* H, float* HD, float* MX,
                      float* DEN, float* AGG, float* SUMS, float* STATS,
                      hipStream_t stream)
{
    const int N = 50000, E = 400000, ED = 64;
    const float *Wsrc = P[0], *bsrc = P[1], *Wdst = P[2], *bdst = P[3];
    const float *We   = P[4], *be   = P[5], *W1   = P[6], *b1   = P[7];
    const float *gam  = P[8], *bet  = P[9], *W2   = P[10], *b2  = P[11];
    const int C = 2 * Fo;
    float* H1 = MX;   // h1 [N, 2*Fo] reuses the contiguous MX+DEN region

    // projections (bf16 WMMA, fp32 accumulate)
    launch_gemm(x,  Wsrc, bsrc, H,   N, Fi, Fo, 0, stream);
    launch_gemm(x,  Wdst, bdst, HD,  N, Fi, Fo, 0, stream);
    launch_gemm(ea, We,   be,   MSG, E, ED, Fo, 0, stream);

    // zero accumulators (after x has been consumed; AGG may alias layer input)
    long long nf4 = (long long)N * Fo / 4;
    unsigned zb = (unsigned)((nf4 + 255) / 256);
    zero_kernel<<<zb, 256, 0, stream>>>(MX,  nf4);
    zero_kernel<<<zb, 256, 0, stream>>>(DEN, nf4);
    zero_kernel<<<zb, 256, 0, stream>>>(AGG, nf4);

    // edge softmax aggregation
    long long em = (long long)E * (Fo / 4);
    edge_message_kernel<<<(unsigned)((em + 255) / 256), 256, 0, stream>>>(MSG, H, src, E, Fo);
    long long ef = (long long)E * Fo;
    unsigned eb = (unsigned)((ef + 255) / 256);
    segmax_kernel<<<eb, 256, 0, stream>>>((unsigned int*)MX, MSG, dst, E, Fo);
    expden_kernel<<<eb, 256, 0, stream>>>(DEN, MSG, (const unsigned int*)MX, dst, E, Fo);
    agg_kernel<<<eb, 256, 0, stream>>>(AGG, MSG, (const unsigned int*)MX, DEN, dst, E, Fo);

    // out = agg + lin_dst(x)  -> into H (h is dead)
    add_kernel<<<zb, 256, 0, stream>>>(H, AGG, HD, nf4);

    // MLP: Linear -> BN(train) -> ReLU -> Linear
    zero_kernel<<<1, 256, 0, stream>>>(SUMS, (long long)(2 * C) / 4);
    launch_gemm(H, W1, b1, H1, N, Fo, C, 0, stream);
    bn_partial_kernel<<<(N + 255) / 256, 256, 0, stream>>>(H1, SUMS, N, C);
    bn_finalize_kernel<<<(C + 255) / 256, 256, 0, stream>>>(SUMS, STATS, N, C);
    long long nc = (long long)N * C;
    bn_apply_relu_kernel<<<(unsigned)((nc + 255) / 256), 256, 0, stream>>>(H1, STATS, gam, bet, nc, C);
    launch_gemm(H1, W2, b2, yout, N, C, Fo, yrelu, stream);
}

extern "C" void kernel_launch(void* const* d_in, const int* in_sizes, int n_in,
                              void* d_out, int out_size, void* d_ws, size_t ws_size,
                              hipStream_t stream) {
    const int N = 50000, E = 400000;
    const float* x  = (const float*)d_in[0];
    const int* eidx = (const int*)d_in[1];
    const float* ea = (const float*)d_in[2];
    const int* src = eidx;
    const int* dst = eidx + E;

    const float* P[24];
    for (int i = 0; i < 24; ++i) P[i] = (const float*)d_in[3 + i];

    // workspace layout (fp32 elements)
    const long long SZ_MSG = (long long)E * 256;   // 102.4M
    const long long SZ_NF  = (long long)N * 256;   // 12.8M
    float* W    = (float*)d_ws;
    float* MSG  = W;
    float* H    = MSG + SZ_MSG;
    float* HD   = H   + SZ_NF;
    float* MX   = HD  + SZ_NF;   // MX+DEN contiguous: doubles as h1 [N, 512]
    float* DEN  = MX  + SZ_NF;
    float* AGG  = DEN + SZ_NF;   // also: layer-1 output / layer-2 input
    float* SUMS = AGG + SZ_NF;
    float* STATS= SUMS + 1024;

    // layer 1: IN=128 -> HID=256, inter-layer relu fused into final GEMM
    run_layer(x, 128, 256, src, dst, ea, P, AGG, /*relu*/1,
              MSG, H, HD, MX, DEN, AGG, SUMS, STATS, stream);
    // layer 2: HID=256 -> OUT=128, writes d_out
    run_layer(AGG, 256, 128, src, dst, ea, P + 12, (float*)d_out, /*relu*/0,
              MSG, H, HD, MX, DEN, AGG, SUMS, STATS, stream);
}
